// BatchContrastiveLoss_44238163148972
// MI455X (gfx1250) — matrix operand order repaired and checked by reading, not compiled, not run on hardware
//
#include <hip/hip_runtime.h>

// Contrastive loss, fully fused for MI455X (gfx1250, wave32, WMMA, async-LDS).
//   Pass 0: init per-row/col stats.
//   Pass 1: normalize rows, split into f16 hi + f16 lo (near-fp32 GEMM via 3 WMMAs).
//   Pass 2: tiled WMMA GEMM with double-buffered ASYNC global->LDS staging;
//           fixed-shift softmax stats (cos/T <= 1/T) + encoded atomicMax argmax + diag.
//   Pass 3: single-block reduction -> 5 outputs.

#define D_DIM 256
#define INV_T 14.285714285714286f   // 1 / 0.07
#define NORM_EPS 1e-8f

// LDS staging layout (per stage): A tiles 128 rows, B tiles 64 rows of 32 halfs
// (64 B) padded to 80 B stride (20*r mod 64 distinct for r=0..15 -> no bank
// conflicts, still 16 B aligned).
#define LDS_STRIDE 80u
#define OFF_AHI 0u
#define OFF_ALO 10240u
#define OFF_BHI 20480u
#define OFF_BLO 25600u
#define STAGE_BYTES 30720u

typedef __attribute__((ext_vector_type(16))) _Float16 v16h;
typedef __attribute__((ext_vector_type(8)))  _Float16 v8h;
typedef __attribute__((ext_vector_type(8)))  float    v8f;
typedef __attribute__((ext_vector_type(4)))  int      v4i;

// Builtin signature (from clang diagnostic): arg0 = int4* in global AS (printed
// as "int __vector(4) __device__ *"), arg1 = LDS side, then imm offset, imm cpol.
typedef __attribute__((address_space(1))) v4i* gv4i_p;
typedef __attribute__((address_space(3))) v4i* lv4i_p;

__device__ __forceinline__ unsigned enc_f32(float f) {
    // order-preserving float -> u32 (for atomicMax over possibly-negative floats)
    unsigned u = __float_as_uint(f);
    return (u & 0x80000000u) ? ~u : (u | 0x80000000u);
}

// ---- gfx1250 async global->LDS copy (ASYNCcnt-tracked), hedged builtin/asm --
__device__ __forceinline__ void async_copy_b128(unsigned lds_off, const void* gsrc) {
#if __has_builtin(__builtin_amdgcn_global_load_async_to_lds_b128)
    __builtin_amdgcn_global_load_async_to_lds_b128(
        (gv4i_p)(size_t)gsrc, (lv4i_p)(size_t)lds_off, 0, 0);
#else
    asm volatile("global_load_async_to_lds_b128 %0, %1, off"
                 :: "v"(lds_off), "v"(gsrc) : "memory");
#endif
}

#if __has_builtin(__builtin_amdgcn_s_wait_asynccnt)
#define WAIT_ASYNC(n) __builtin_amdgcn_s_wait_asynccnt(n)
#else
#define WAIT_ASYNC(n) asm volatile("s_wait_asynccnt %0" :: "i"(n) : "memory")
#endif

// ---- WMMA fragment loaders from LDS (layouts per CDNA5 ISA 7.12.2) ----------
// A: 16x32 f16. lane m in [0,16): row m, K {0..7,16..23} -> bytes {0..15,32..47};
// lanes 16..31: K {8..15,24..31} -> bytes {16..31,48..63}.
__device__ __forceinline__ v16h lds_a_frag(const char* base, int mrel, int lane) {
    const char* p = base + (unsigned)(mrel + (lane & 15)) * LDS_STRIDE + ((lane >> 4) << 4);
    v8h x = *(const v8h*)p;
    v8h y = *(const v8h*)(p + 32);
    return __builtin_shufflevector(x, y, 0,1,2,3,4,5,6,7,8,9,10,11,12,13,14,15);
}

// B: 32x16 f16 = G^T tile; lane n in [0,16): column n = G row n, K {0..7,16..23}
// (bytes {0..15,32..47}); lanes 16..31: K {8..15,24..31} (bytes {16..31,48..63}).
__device__ __forceinline__ v16h lds_b_frag(const char* base, int nrel, int lane) {
    const char* p = base + (unsigned)(nrel + (lane & 15)) * LDS_STRIDE + ((lane >> 4) << 4);
    v8h x = *(const v8h*)p;
    v8h y = *(const v8h*)(p + 32);
    return __builtin_shufflevector(x, y, 0,1,2,3,4,5,6,7,8,9,10,11,12,13,14,15);
}

// ---- Pass 0: init stats -----------------------------------------------------
__global__ void init_stats_kernel(float* rowsum, float* colsum,
                                  unsigned* rowmax, unsigned* colmax,
                                  float* diag, int B) {
    int i = blockIdx.x * blockDim.x + threadIdx.x;
    if (i < B) {
        rowsum[i] = 0.f; colsum[i] = 0.f;
        rowmax[i] = 0u;  colmax[i] = 0u;   // enc(x) > 0 for all finite x
        diag[i]   = 0.f;
    }
}

// ---- Pass 1: normalize + f16 hi/lo split (one wave per row) -----------------
__global__ __launch_bounds__(256)
void normalize_split_kernel(const float* __restrict__ T, const float* __restrict__ G,
                            _Float16* __restrict__ Thi, _Float16* __restrict__ Tlo,
                            _Float16* __restrict__ Ghi, _Float16* __restrict__ Glo,
                            int B) {
    const int lane = threadIdx.x & 31;
    const int wave = threadIdx.x >> 5;
    const int row  = blockIdx.x * 8 + wave;   // 0..2B-1

    const float* src; _Float16 *dhi, *dlo; int r;
    if (row < B) { src = T; dhi = Thi; dlo = Tlo; r = row; }
    else         { src = G; dhi = Ghi; dlo = Glo; r = row - B; }

    const float4* p = (const float4*)(src + (size_t)r * D_DIM) + lane * 2;
    float4 x0 = p[0], x1 = p[1];
    float v[8] = {x0.x, x0.y, x0.z, x0.w, x1.x, x1.y, x1.z, x1.w};

    float ss = 0.f;
#pragma unroll
    for (int i = 0; i < 8; ++i) ss += v[i] * v[i];
#pragma unroll
    for (int off = 1; off < 32; off <<= 1) ss += __shfl_xor(ss, off, 32);

    float scale = 1.f / fmaxf(sqrtf(ss), NORM_EPS);

    v8h hi, lo;
#pragma unroll
    for (int i = 0; i < 8; ++i) {
        float y = v[i] * scale;
        _Float16 h = (_Float16)y;
        hi[i] = h;
        lo[i] = (_Float16)(y - (float)h);
    }
    *(v8h*)(dhi + (size_t)r * D_DIM + lane * 8) = hi;
    *(v8h*)(dlo + (size_t)r * D_DIM + lane * 8) = lo;
}

// ---- Pass 2: fused WMMA GEMM + softmax/argmax stats -------------------------
// Block = 8 waves (256 thr): block tile 128(M) x 64(N); wave tile 32x32.
// Stage = 1536 b128 chunks (A hi/lo 512 ea, B hi/lo 256 ea) = 6 async/thread.
__device__ __forceinline__ void stage_copy(
        const _Float16* Thi, const _Float16* Tlo,
        const _Float16* Ghi, const _Float16* Glo,
        int Rblk, int Cblk, int k0, unsigned bufBase, int tid) {
#pragma unroll
    for (int h = 0; h < 2; ++h) {
        int id = tid + h * 256;
        int row = id >> 2, c = id & 3;
        size_t goff = (size_t)(Rblk + row) * D_DIM + k0 + c * 8;
        unsigned loff = row * LDS_STRIDE + c * 16u;
        async_copy_b128(bufBase + OFF_AHI + loff, Thi + goff);
        async_copy_b128(bufBase + OFF_ALO + loff, Tlo + goff);
    }
    {
        int row = tid >> 2, c = tid & 3;
        size_t goff = (size_t)(Cblk + row) * D_DIM + k0 + c * 8;
        unsigned loff = row * LDS_STRIDE + c * 16u;
        async_copy_b128(bufBase + OFF_BHI + loff, Ghi + goff);
        async_copy_b128(bufBase + OFF_BLO + loff, Glo + goff);
    }
}

__global__ __launch_bounds__(256)
void gemm_stats_kernel(const _Float16* __restrict__ Thi, const _Float16* __restrict__ Tlo,
                       const _Float16* __restrict__ Ghi, const _Float16* __restrict__ Glo,
                       float* __restrict__ rowsum, float* __restrict__ colsum,
                       unsigned* __restrict__ rowmax, unsigned* __restrict__ colmax,
                       float* __restrict__ diag, int B) {
    __shared__ char smem[2 * STAGE_BYTES];   // 60 KB of 320 KB/WGP

    const int tid  = threadIdx.x;
    const int lane = tid & 31;
    const int wave = tid >> 5;
    const int wm = wave >> 1, wn = wave & 1;
    const int Rblk = blockIdx.y * 128;
    const int Cblk = blockIdx.x * 64;
    const int Rbase = Rblk + wm * 32;
    const int Cbase = Cblk + wn * 32;

    // AS3 pointer value == LDS byte offset == low 32 bits of generic pointer.
    const unsigned smemBase = (unsigned)(size_t)&smem[0];

    stage_copy(Thi, Tlo, Ghi, Glo, Rblk, Cblk, 0, smemBase, tid);

    v8f acc[2][2] = {};

    for (int kt = 0; kt < D_DIM / 32; ++kt) {
        if (kt + 1 < D_DIM / 32) {
            stage_copy(Thi, Tlo, Ghi, Glo, Rblk, Cblk, (kt + 1) * 32,
                       smemBase + ((kt + 1) & 1) * STAGE_BYTES, tid);
            WAIT_ASYNC(6);   // async loads complete in order -> stage kt landed
        } else {
            WAIT_ASYNC(0);
        }
        __syncthreads();     // all waves' stage-kt data visible

        const char* buf = smem + (kt & 1) * STAGE_BYTES;
        v16h ah[2], al[2], bh[2], bl[2];
#pragma unroll
        for (int mi = 0; mi < 2; ++mi) {
            ah[mi] = lds_a_frag(buf + OFF_AHI, wm * 32 + mi * 16, lane);
            al[mi] = lds_a_frag(buf + OFF_ALO, wm * 32 + mi * 16, lane);
        }
#pragma unroll
        for (int ni = 0; ni < 2; ++ni) {
            bh[ni] = lds_b_frag(buf + OFF_BHI, wn * 32 + ni * 16, lane);
            bl[ni] = lds_b_frag(buf + OFF_BLO, wn * 32 + ni * 16, lane);
        }
#pragma unroll
        for (int mi = 0; mi < 2; ++mi)
#pragma unroll
            for (int ni = 0; ni < 2; ++ni) {
                // split-precision product: th*gh + th*gl + tl*gh (~fp32 accurate)
                acc[mi][ni] = __builtin_amdgcn_wmma_f32_16x16x32_f16(
                    false, ah[mi], false, bh[ni], (short)0, acc[mi][ni], false, false);
                acc[mi][ni] = __builtin_amdgcn_wmma_f32_16x16x32_f16(
                    false, ah[mi], false, bl[ni], (short)0, acc[mi][ni], false, false);
                acc[mi][ni] = __builtin_amdgcn_wmma_f32_16x16x32_f16(
                    false, al[mi], false, bh[ni], (short)0, acc[mi][ni], false, false);
            }
        __syncthreads();     // all reads of buf done before it is re-staged
    }

    // scale to logits; fixed shift M = INV_T bounds exp in [e^-28.6, ~1]
    float ev[2][2][8];
#pragma unroll
    for (int mi = 0; mi < 2; ++mi)
#pragma unroll
        for (int ni = 0; ni < 2; ++ni)
#pragma unroll
            for (int r = 0; r < 8; ++r) {
                float s = acc[mi][ni][r] * INV_T;
                acc[mi][ni][r] = s;
                ev[mi][ni][r] = __expf(s - INV_T);
            }

    // Row stats: row = Rbase + mi*16 + 8*(lane>=16) + r; 16 cols live in one
    // 16-lane group -> xor-shuffle reduce, one atomic per row per wave.
#pragma unroll
    for (int mi = 0; mi < 2; ++mi)
#pragma unroll
        for (int r = 0; r < 8; ++r) {
            float es = ev[mi][0][r] + ev[mi][1][r];
            float ms = fmaxf(acc[mi][0][r], acc[mi][1][r]);
#pragma unroll
            for (int off = 1; off < 16; off <<= 1) {
                es += __shfl_xor(es, off, 32);
                ms = fmaxf(ms, __shfl_xor(ms, off, 32));
            }
            if ((lane & 15) == 0) {
                int row = Rbase + mi * 16 + ((lane >> 4) << 3) + r;
                atomicAdd(&rowsum[row], es);
                atomicMax(&rowmax[row], enc_f32(ms));
            }
        }

    // Col stats: col = Cbase + ni*16 + (lane&15); 16 rows = 8 regs x 2 lane groups.
#pragma unroll
    for (int ni = 0; ni < 2; ++ni) {
        float ec = 0.f, mc = -3.0e38f;
#pragma unroll
        for (int mi = 0; mi < 2; ++mi)
#pragma unroll
            for (int r = 0; r < 8; ++r) {
                ec += ev[mi][ni][r];
                mc = fmaxf(mc, acc[mi][ni][r]);
            }
        ec += __shfl_xor(ec, 16, 32);
        mc = fmaxf(mc, __shfl_xor(mc, 16, 32));
        if (lane < 16) {
            int col = Cbase + ni * 16 + lane;
            atomicAdd(&colsum[col], ec);
            atomicMax(&colmax[col], enc_f32(mc));
        }
    }

    // Diagonal: element (m,n)=(r+8*(lane>=16), lane&15); m==n hits lanes 0..7
    // (r==lane) and 24..31 (r==lane-24).
#pragma unroll
    for (int mi = 0; mi < 2; ++mi)
#pragma unroll
        for (int ni = 0; ni < 2; ++ni)
            if (Rbase + mi * 16 == Cbase + ni * 16) {
#pragma unroll
                for (int r = 0; r < 8; ++r) {
                    bool isDiag = (lane < 8 && lane == r) || (lane >= 24 && lane - 24 == r);
                    if (isDiag)
                        diag[Rbase + mi * 16 + (lane & 15)] = acc[mi][ni][r];
                }
            }
}

// ---- Pass 3: final reduction -> [total, t2g, g2t, t2g_acc, g2t_acc] ---------
__global__ __launch_bounds__(256)
void finalize_kernel(const float* __restrict__ rowsum, const float* __restrict__ colsum,
                     const unsigned* __restrict__ rowmax, const unsigned* __restrict__ colmax,
                     const float* __restrict__ diag, float* __restrict__ out, int B) {
    const int lane = threadIdx.x & 31;
    const int wave = threadIdx.x >> 5;
    float t2g = 0.f, g2t = 0.f, ta = 0.f, ga = 0.f;

    for (int i = threadIdx.x; i < B; i += 256) {
        float d = diag[i];
        t2g += (INV_T + logf(rowsum[i]) - d);   // -(d - logsumexp_row)
        g2t += (INV_T + logf(colsum[i]) - d);
        unsigned de = enc_f32(d);
        ta += (de == rowmax[i]) ? 1.f : 0.f;    // diag is the row max (bit-exact)
        ga += (de == colmax[i]) ? 1.f : 0.f;
    }
#pragma unroll
    for (int off = 1; off < 32; off <<= 1) {
        t2g += __shfl_xor(t2g, off, 32);
        g2t += __shfl_xor(g2t, off, 32);
        ta  += __shfl_xor(ta,  off, 32);
        ga  += __shfl_xor(ga,  off, 32);
    }
    __shared__ float sm[4][8];
    if (lane == 0) { sm[0][wave] = t2g; sm[1][wave] = g2t; sm[2][wave] = ta; sm[3][wave] = ga; }
    __syncthreads();
    if (threadIdx.x == 0) {
        float a = 0.f, b = 0.f, c = 0.f, d = 0.f;
        for (int w = 0; w < 8; ++w) { a += sm[0][w]; b += sm[1][w]; c += sm[2][w]; d += sm[3][w]; }
        float m = 1.f / (float)B;
        out[1] = a * m; out[2] = b * m; out[0] = out[1] + out[2];
        out[3] = c * m; out[4] = d * m;
    }
}

extern "C" void kernel_launch(void* const* d_in, const int* in_sizes, int n_in,
                              void* d_out, int out_size, void* d_ws, size_t ws_size,
                              hipStream_t stream) {
    const float* T = (const float*)d_in[0];
    const float* G = (const float*)d_in[1];
    const int B = in_sizes[0] / D_DIM;   // 8192

    char* ws = (char*)d_ws;
    size_t off = 0;
    const size_t embBytes = (size_t)B * D_DIM * sizeof(_Float16);
    _Float16* Thi = (_Float16*)(ws + off); off += embBytes;
    _Float16* Tlo = (_Float16*)(ws + off); off += embBytes;
    _Float16* Ghi = (_Float16*)(ws + off); off += embBytes;
    _Float16* Glo = (_Float16*)(ws + off); off += embBytes;
    float*    rowsum = (float*)(ws + off);    off += (size_t)B * 4;
    float*    colsum = (float*)(ws + off);    off += (size_t)B * 4;
    unsigned* rowmax = (unsigned*)(ws + off); off += (size_t)B * 4;
    unsigned* colmax = (unsigned*)(ws + off); off += (size_t)B * 4;
    float*    diag   = (float*)(ws + off);    off += (size_t)B * 4;
    float* out = (float*)d_out;

    init_stats_kernel<<<(B + 255) / 256, 256, 0, stream>>>(rowsum, colsum, rowmax, colmax, diag, B);
    normalize_split_kernel<<<(2 * B + 7) / 8, 256, 0, stream>>>(T, G, Thi, Tlo, Ghi, Glo, B);
    dim3 grid(B / 64, B / 128);   // N tiles of 64, M tiles of 128
    gemm_stats_kernel<<<grid, 256, 0, stream>>>(Thi, Tlo, Ghi, Glo,
                                                rowsum, colsum, rowmax, colmax, diag, B);
    finalize_kernel<<<1, 256, 0, stream>>>(rowsum, colsum, rowmax, colmax, diag, out, B);
}
